// Module_1_14087492731433
// MI455X (gfx1250) — compile-verified, matrix operand order checked by reading, not run on hardware
//
#include <hip/hip_runtime.h>
#include <hip/hip_bf16.h>

// ---------------------------------------------------------------------------
// Problem dims (fixed by reference setup_inputs)
// ---------------------------------------------------------------------------
#define B_DIM 256
#define T_DIM 240
#define N_DIM 264
#define H_DIM 256
#define NP    288   // N padded: multiple of 32 (valid M-tile and K dim)
#define TP    256   // T padded: multiple of 32

typedef __bf16 bf16_t;
typedef __bf16 bf16x8  __attribute__((ext_vector_type(8)));
typedef __bf16 bf16x16 __attribute__((ext_vector_type(16)));
typedef float  f32x8   __attribute__((ext_vector_type(8)));

// ---------------------------------------------------------------------------
// Fragment load: 16x32 bf16 operand per wave from a row-major source.
// CDNA5 16-bit A-matrix 16x32 layout:
//   lanes 0-15  : M = lane,    VGPR0..3 = K 0..7,  VGPR4..7 = K 16..23
//   lanes 16-31 : M = lane-16, VGPR0..3 = K 8..15, VGPR4..7 = K 24..31
// => per lane: two contiguous 16-byte chunks. B operand loads identically
// from Bt (row n of Bt == column n of B).
// ---------------------------------------------------------------------------
__device__ __forceinline__ bf16x16 ldfrag(const bf16_t* __restrict__ p) {
  bf16x8 lo = *(const bf16x8*)(p);
  bf16x8 hi = *(const bf16x8*)(p + 16);
  return __builtin_shufflevector(lo, hi, 0,1,2,3,4,5,6,7,8,9,10,11,12,13,14,15);
}

#define WMMA_BF16(A_, B_, C_) \
  __builtin_amdgcn_wmma_f32_16x16x32_bf16(false, (A_), false, (B_), (short)0, (C_), false, false)

// ---------------------------------------------------------------------------
// Batched GEMM: C[M,Nsz] = A[M,K] * Bt[Nsz,K]^T   (bf16 in, f32 accumulate)
//   MODE 0: store f32 row-major            (cov)
//   MODE 1: store f32 row-major + bias[n]  (x1/x2 pre-activations)
//   MODE 2: store bf16 TRANSPOSED Ct[n,m]  (feeds B-side of next GEMM)
// One wave computes a 32x64 tile (2 M-tiles x 4 N-tiles, 8 accumulators);
// block = (32,4) waves => 32x256. K = KITER*32 (compile-time).
// GUARD=false: grid covers Nsz exactly -> straight-line, branch-free code.
// GUARD=true : scalar (SGPR) guards via readfirstlane -> EXEC untouched,
//              WMMA always runs with EXEC all-1s as the ISA requires.
// ---------------------------------------------------------------------------
template<int MODE, int KITER, bool GUARD>
__launch_bounds__(128)
__global__ void gemm_wmma(const bf16_t* __restrict__ A,  unsigned long long sA, int lda,
                          const bf16_t* __restrict__ Bt, unsigned long long sB, int ldb,
                          void* __restrict__ Cv,         unsigned long long sC, int ldc,
                          const float* __restrict__ bias,
                          int Nsz)
{
  const int b    = blockIdx.z;
  const int lane = threadIdx.x & 31;
  const int m0   = blockIdx.y * 32;
  // Wave-uniform by construction (wave = 32 threads along x); force to SGPR so
  // all guards compile to scalar branches (no EXEC manipulation around WMMA).
  const int nbase = __builtin_amdgcn_readfirstlane((blockIdx.x * 4 + threadIdx.y) * 64);
  if (GUARD && nbase >= Nsz) return;

  const bool v1 = !GUARD || (nbase + 16) < Nsz;
  const bool v2 = !GUARD || (nbase + 32) < Nsz;
  const bool v3 = !GUARD || (nbase + 48) < Nsz;

  const int rsel = lane & 15;
  const int ksel = (lane >> 4) << 3;

  const bf16_t* pa0 = A + (size_t)b * sA + (size_t)(m0 + rsel) * lda + ksel;
  const bf16_t* pa1 = pa0 + (size_t)16 * lda;
  const bf16_t* Btb = Bt + (size_t)b * sB;
  const bf16_t* pb0 = Btb + (size_t)(nbase +  0 + rsel) * ldb + ksel;
  const bf16_t* pb1 = Btb + (size_t)(nbase + 16 + rsel) * ldb + ksel;
  const bf16_t* pb2 = Btb + (size_t)(nbase + 32 + rsel) * ldb + ksel;
  const bf16_t* pb3 = Btb + (size_t)(nbase + 48 + rsel) * ldb + ksel;

  f32x8 acc00 = {}, acc01 = {}, acc02 = {}, acc03 = {};
  f32x8 acc10 = {}, acc11 = {}, acc12 = {}, acc13 = {};

  for (int ki = 0; ki < KITER; ++ki) {
    bf16x16 a0 = ldfrag(pa0);
    bf16x16 a1 = ldfrag(pa1);
    pa0 += 32; pa1 += 32;
    {
      bf16x16 bb = ldfrag(pb0); pb0 += 32;
      acc00 = WMMA_BF16(a0, bb, acc00);
      acc10 = WMMA_BF16(a1, bb, acc10);
    }
    if (v1) {
      bf16x16 bb = ldfrag(pb1); pb1 += 32;
      acc01 = WMMA_BF16(a0, bb, acc01);
      acc11 = WMMA_BF16(a1, bb, acc11);
    }
    if (v2) {
      bf16x16 bb = ldfrag(pb2); pb2 += 32;
      acc02 = WMMA_BF16(a0, bb, acc02);
      acc12 = WMMA_BF16(a1, bb, acc12);
    }
    if (v3) {
      bf16x16 bb = ldfrag(pb3); pb3 += 32;
      acc03 = WMMA_BF16(a0, bb, acc03);
      acc13 = WMMA_BF16(a1, bb, acc13);
    }
  }

  // C/D layout: lane<16 -> N=lane,   M = mtile + r  (r = acc VGPR index 0..7)
  //             lane>=16-> N=lane-16,M = mtile + 8 + r
  const int nl  = lane & 15;
  const int mr0 = m0 + ((lane >> 4) << 3);
  const int mr1 = mr0 + 16;

#define ST_TILE(ACC, NT, MR) do {                                              \
    const int n = nbase + (NT) * 16 + nl;                                      \
    if (MODE == 2) {                                                           \
      bf16_t* Cb = (bf16_t*)Cv + (size_t)b * sC;                               \
      bf16x8 ov;                                                               \
      _Pragma("unroll") for (int r = 0; r < 8; ++r) ov[r] = (bf16_t)(ACC)[r];  \
      *(bf16x8*)(Cb + (size_t)n * ldc + (MR)) = ov;                            \
    } else {                                                                   \
      float* Cb = (float*)Cv + (size_t)b * sC;                                 \
      const float bv = (MODE == 1) ? bias[n] : 0.0f;                           \
      _Pragma("unroll") for (int r = 0; r < 8; ++r)                            \
        Cb[(size_t)((MR) + r) * ldc + n] = (ACC)[r] + bv;                      \
    }                                                                          \
  } while (0)

  ST_TILE(acc00, 0, mr0); ST_TILE(acc10, 0, mr1);
  if (v1) { ST_TILE(acc01, 1, mr0); ST_TILE(acc11, 1, mr1); }
  if (v2) { ST_TILE(acc02, 2, mr0); ST_TILE(acc12, 2, mr1); }
  if (v3) { ST_TILE(acc03, 3, mr0); ST_TILE(acc13, 3, mr1); }
#undef ST_TILE
}

// ---------------------------------------------------------------------------
// data [B,T,N] f32 -> xmT [B,NP,TP] bf16: per-ROI mean removal, transpose,
// zero-pad (zeros are inert in cov = xmT * xmT^T).
// ---------------------------------------------------------------------------
__global__ void center_transpose(const float* __restrict__ data, bf16_t* __restrict__ xmT)
{
  const int b = blockIdx.x;
  const float* db = data + (size_t)b * T_DIM * N_DIM;
  bf16_t* xb = xmT + (size_t)b * NP * TP;
  for (int n = threadIdx.x; n < NP; n += blockDim.x) {
    if (n < N_DIM) {
      float s = 0.f;
      for (int t = 0; t < T_DIM; ++t) s += db[(size_t)t * N_DIM + n];
      const float mean = s * (1.0f / T_DIM);
      for (int t = 0; t < T_DIM; ++t)
        xb[(size_t)n * TP + t] = (bf16_t)(db[(size_t)t * N_DIM + n] - mean);
      for (int t = T_DIM; t < TP; ++t) xb[(size_t)n * TP + t] = (bf16_t)0.0f;
    } else {
      for (int t = 0; t < TP; ++t) xb[(size_t)n * TP + t] = (bf16_t)0.0f;
    }
  }
}

// W [K,H] f32 -> Wt [H,KP] bf16 (transposed + K zero-padded)
__global__ void transpose_w(const float* __restrict__ W, bf16_t* __restrict__ Wt,
                            int K, int KP)
{
  const int h = blockIdx.x;               // 0..H-1
  for (int k = threadIdx.x; k < KP; k += blockDim.x) {
    float v = (k < K) ? W[(size_t)k * H_DIM + h] : 0.0f;
    Wt[(size_t)h * KP + k] = (bf16_t)v;
  }
}

// d[b,i] = sqrt(cov[b,i,i])
__global__ void diag_kernel(const float* __restrict__ cov, float* __restrict__ dvec)
{
  const int b = blockIdx.x, i = threadIdx.x;     // blockDim = NP
  float v = 0.f;
  if (i < N_DIM)
    v = sqrtf(fmaxf(cov[(size_t)b * NP * NP + (size_t)i * NP + i], 0.0f));
  dvec[(size_t)b * NP + i] = v;
}

// adj_ij = |nan_to_num(cov_ij/(d_i d_j))|; emit padded bf16 + unpadded f32
// output; accumulate rowsum(adj + I) for the symmetric normalization.
__global__ void adj_rowsum(const float* __restrict__ cov, const float* __restrict__ dvec,
                           bf16_t* __restrict__ adjp, float* __restrict__ adj_out,
                           float* __restrict__ rowsum)
{
  const int b = blockIdx.y, i = blockIdx.x;      // grid (NP, B), block 128
  __shared__ float red[4];
  const float di = dvec[(size_t)b * NP + i];
  float lsum = 0.f;
  for (int j = threadIdx.x; j < NP; j += blockDim.x) {
    float v = 0.f;
    if (i < N_DIM && j < N_DIM) {
      const float dj = dvec[(size_t)b * NP + j];
      float corr = cov[(size_t)b * NP * NP + (size_t)i * NP + j] / (di * dj);
      if (!(fabsf(corr) < 3.0e38f)) corr = 0.0f;  // nan/inf -> 0
      v = fabsf(corr);
      adj_out[(size_t)b * N_DIM * N_DIM + (size_t)i * N_DIM + j] = v;
    }
    adjp[(size_t)b * NP * NP + (size_t)i * NP + j] = (bf16_t)v;
    lsum += v + ((i == j && i < N_DIM) ? 1.0f : 0.0f);
  }
  #pragma unroll
  for (int off = 16; off > 0; off >>= 1) lsum += __shfl_down(lsum, off, 32);
  if ((threadIdx.x & 31) == 0) red[threadIdx.x >> 5] = lsum;
  __syncthreads();
  if (threadIdx.x == 0)
    rowsum[(size_t)b * NP + i] = red[0] + red[1] + red[2] + red[3];
}

__global__ void dinv_kernel(const float* __restrict__ rowsum, float* __restrict__ dinv)
{
  const int b = blockIdx.x, i = threadIdx.x;     // blockDim = NP
  float v = 0.f;
  if (i < N_DIM) {
    const float s = rowsum[(size_t)b * NP + i];
    v = (s > 0.f) ? rsqrtf(s) : 0.f;
  }
  dinv[(size_t)b * NP + i] = v;
}

// L_ij = dinv_i * (adj_ij + I_ij) * dinv_j, padded bf16 (pad rows/cols = 0 so
// garbage in activation pad rows cannot leak through K of L @ Y).
__global__ void build_L(const bf16_t* __restrict__ adjp, const float* __restrict__ dinv,
                        bf16_t* __restrict__ Lp)
{
  const int b = blockIdx.y, i = blockIdx.x;      // grid (NP, B), block 128
  const float di = dinv[(size_t)b * NP + i];
  for (int j = threadIdx.x; j < NP; j += blockDim.x) {
    float v = 0.f;
    if (i < N_DIM && j < N_DIM) {
      const float a = (float)adjp[(size_t)b * NP * NP + (size_t)i * NP + j];
      v = di * (a + ((i == j) ? 1.0f : 0.0f)) * dinv[(size_t)b * NP + j];
    }
    Lp[(size_t)b * NP * NP + (size_t)i * NP + j] = (bf16_t)v;
  }
}

// Row-wise hyperbolic fkernel (project -> logmap0) + Cosine activation.
// FINAL=false: write bf16 g1 [B,NP,H] (A operand of g1@W2)
// FINAL=true : write f32 g2 unpadded to d_out
template<bool FINAL>
__global__ void fkernel_act(const float* __restrict__ x, bf16_t* __restrict__ gbf,
                            float* __restrict__ gout)
{
  const int b = blockIdx.y;
  const int i = blockIdx.x;
  const int h = threadIdx.x;                     // blockDim = H = 256 (8 waves)
  const float v = x[((size_t)b * NP + i) * H_DIM + h];
  __shared__ float red[8];
  float s = v * v;
  #pragma unroll
  for (int off = 16; off > 0; off >>= 1) s += __shfl_down(s, off, 32);
  if ((h & 31) == 0) red[h >> 5] = s;
  __syncthreads();
  const float tot  = red[0] + red[1] + red[2] + red[3] +
                     red[4] + red[5] + red[6] + red[7];
  const float norm    = fmaxf(sqrtf(tot), 1e-15f);
  const float maxnorm = 1.0f - 4e-3f;            // (1-eps)/sqrt(c), c=1
  const float scale   = (norm > maxnorm) ? (maxnorm / norm) : 1.0f;
  const float pn      = fmaxf(norm * scale, 1e-15f);
  const float g       = (atanhf(pn) / pn) * scale * v;   // logmap0(project(x))
  const float outv    = fmaxf(g, 0.0f) + 0.5f * cosf(g + 0.94247778f); // relu + a*cos(x+phi)
  if (FINAL) {
    gout[((size_t)b * N_DIM + i) * H_DIM + h] = outv;
  } else {
    gbf[((size_t)b * NP + i) * H_DIM + h] = (bf16_t)outv;
  }
}

// ---------------------------------------------------------------------------
// Orchestration
// ---------------------------------------------------------------------------
extern "C" void kernel_launch(void* const* d_in, const int* in_sizes, int n_in,
                              void* d_out, int out_size, void* d_ws, size_t ws_size,
                              hipStream_t stream)
{
  (void)in_sizes; (void)n_in; (void)out_size; (void)ws_size;
  const float* data = (const float*)d_in[0];     // [B,T,N]
  const float* W1   = (const float*)d_in[1];     // [N,H]
  const float* b1   = (const float*)d_in[2];     // [H]
  const float* W2   = (const float*)d_in[3];     // [H,H]
  const float* b2   = (const float*)d_in[4];     // [H]

  float* g2_out  = (float*)d_out;                                        // [B,N,H]
  float* adj_out = (float*)d_out + (size_t)B_DIM * N_DIM * H_DIM;        // [B,N,N]

  // Workspace carve-up (all offsets 256B aligned).
  char* ws = (char*)d_ws;
  size_t o = 0;
  bf16_t* xmT  = (bf16_t*)(ws + o); o += (size_t)B_DIM * NP * TP * 2;    // reused as g1
  float*  cov  = (float*) (ws + o); o += (size_t)B_DIM * NP * NP * 4;    // reused as x1/x2
  bf16_t* adjp = (bf16_t*)(ws + o); o += (size_t)B_DIM * NP * NP * 2;
  bf16_t* Lp   = (bf16_t*)(ws + o); o += (size_t)B_DIM * NP * NP * 2;
  bf16_t* Yt   = (bf16_t*)(ws + o); o += (size_t)B_DIM * H_DIM * NP * 2;
  bf16_t* W1t  = (bf16_t*)(ws + o); o += (size_t)H_DIM * NP * 2;
  bf16_t* W2t  = (bf16_t*)(ws + o); o += (size_t)H_DIM * H_DIM * 2;
  float*  dvec = (float*) (ws + o); o += (size_t)B_DIM * NP * 4;
  float*  rsum = (float*) (ws + o); o += (size_t)B_DIM * NP * 4;
  float*  dinv = (float*) (ws + o); o += (size_t)B_DIM * NP * 4;
  bf16_t* g1   = xmT;                            // xmT is dead after cov GEMM
  float*  xact = cov;                            // cov is dead after build_L

  const dim3 gblk(32, 4);
  const unsigned long long sNN = (unsigned long long)NP * NP;
  const unsigned long long sNT = (unsigned long long)NP * TP;
  const unsigned long long sNH = (unsigned long long)NP * H_DIM;
  const unsigned long long sHN = (unsigned long long)H_DIM * NP;

  // 1) center + transpose + pad
  center_transpose<<<dim3(B_DIM), dim3(256), 0, stream>>>(data, xmT);
  // 2) weight transposes (bf16)
  transpose_w<<<dim3(H_DIM), dim3(256), 0, stream>>>(W1, W1t, N_DIM, NP);
  transpose_w<<<dim3(H_DIM), dim3(256), 0, stream>>>(W2, W2t, H_DIM, H_DIM);
  // 3) cov = xmT @ xmT^T   (M=NP, N=NP=288 -> guarded, K=TP=256 -> KITER=8)
  gemm_wmma<0, 8, true><<<dim3(2, NP / 32, B_DIM), gblk, 0, stream>>>(
      xmT, sNT, TP, xmT, sNT, TP, cov, sNN, NP, nullptr, NP);
  // 4) correlation -> adj (+ output), row sums, D^-1/2, L
  diag_kernel<<<dim3(B_DIM), dim3(NP), 0, stream>>>(cov, dvec);
  adj_rowsum<<<dim3(NP, B_DIM), dim3(128), 0, stream>>>(cov, dvec, adjp, adj_out, rsum);
  dinv_kernel<<<dim3(B_DIM), dim3(NP), 0, stream>>>(rsum, dinv);
  build_L<<<dim3(NP, B_DIM), dim3(128), 0, stream>>>(adjp, dinv, Lp);
  // 5) Y1^T = (adj @ W1)^T   (N=256 exact -> branch-free, K=NP=288 -> KITER=9)
  gemm_wmma<2, 9, false><<<dim3(1, NP / 32, B_DIM), gblk, 0, stream>>>(
      adjp, sNN, NP, W1t, 0ull, NP, Yt, sHN, NP, nullptr, H_DIM);
  // 6) x1 = L @ Y1 + b1      (N=256 exact, K=288)
  gemm_wmma<1, 9, false><<<dim3(1, NP / 32, B_DIM), gblk, 0, stream>>>(
      Lp, sNN, NP, Yt, sHN, NP, xact, sNH, H_DIM, b1, H_DIM);
  // 7) g1 = act(fkernel(x1)) -> bf16
  fkernel_act<false><<<dim3(NP, B_DIM), dim3(H_DIM), 0, stream>>>(xact, g1, nullptr);
  // 8) Y2^T = (g1 @ W2)^T    (N=256 exact, K=256 -> KITER=8)
  gemm_wmma<2, 8, false><<<dim3(1, NP / 32, B_DIM), gblk, 0, stream>>>(
      g1, sNH, H_DIM, W2t, 0ull, H_DIM, Yt, sHN, NP, nullptr, H_DIM);
  // 9) x2 = L @ Y2 + b2      (N=256 exact, K=288)
  gemm_wmma<1, 9, false><<<dim3(1, NP / 32, B_DIM), gblk, 0, stream>>>(
      Lp, sNN, NP, Yt, sHN, NP, xact, sNH, H_DIM, b2, H_DIM);
  // 10) g2 = act(fkernel(x2)) -> f32 output (unpadded)
  fkernel_act<true><<<dim3(N_DIM, B_DIM), dim3(H_DIM), 0, stream>>>(xact, nullptr, g2_out);
}
